// WienerDeblurNet_26929444946037
// MI455X (gfx1250) — matrix operand order repaired and checked by reading, not compiled, not run on hardware
//
#include <hip/hip_runtime.h>
#include <hip/hip_bf16.h>

typedef __attribute__((ext_vector_type(16))) _Float16 v16h;
typedef __attribute__((ext_vector_type(8)))  float    v8f;
typedef unsigned int u32;

constexpr int   HP    = 270;
constexpr int   PLANE = HP * HP;          // 72900
constexpr float PI2   = 6.28318530717958647692f;

__device__ __forceinline__ int reflectI(int i, int n) {
  if (i < 0)  i = -1 - i;
  if (i >= n) i = 2 * n - 1 - i;
  return i;
}

__device__ __forceinline__ u32 packh2(float a, float b) {
  union { _Float16 h[2]; u32 u; } p;
  p.h[0] = (_Float16)a; p.h[1] = (_Float16)b;
  return p.u;
}

__device__ __forceinline__ float2 cmulc(float2 a, float2 b) {
  return make_float2(a.x * b.x - a.y * b.y, a.x * b.y + a.y * b.x);
}

// ---------------------------------------------------------------------------
// Symmetric pad 256 -> 270 (pad 7 each side)
// ---------------------------------------------------------------------------
__global__ __launch_bounds__(256) void k_sympad(const float* __restrict__ x,
                                                float* __restrict__ xp) {
  int i = blockIdx.x * 256 + threadIdx.x;
  if (i >= 2 * PLANE) return;
  int b = i / PLANE, r = i - b * PLANE;
  int h = r / HP, w = r - h * HP;
  int sh = h - 7; if (sh < 0) sh = -1 - sh; if (sh >= 256) sh = 511 - sh;
  int sw = w - 7; if (sw < 0) sw = -1 - sw; if (sw >= 256) sw = 511 - sw;
  xp[i] = x[(size_t)b * 65536 + sh * 256 + sw];
}

// ---------------------------------------------------------------------------
// Edgetaper 1D profiles ah (H) and aw (W): a1(psf.sum(axis), 270)
// ---------------------------------------------------------------------------
__global__ __launch_bounds__(256) void k_et1d(const float* __restrict__ bk,
                                              float* __restrict__ ah,
                                              float* __restrict__ aw) {
  __shared__ float p[2][15];
  __shared__ float pw[2][270];
  __shared__ float zz[2][270];
  __shared__ float mx[2];
  int t = threadIdx.x;
  if (t < 15) {            // row sums -> axis for H
    float s = 0.f; for (int j = 0; j < 15; ++j) s += bk[t * 15 + j];
    p[0][t] = s;
  } else if (t < 30) {     // col sums -> axis for W
    int c = t - 15; float s = 0.f; for (int j = 0; j < 15; ++j) s += bk[j * 15 + c];
    p[1][c] = s;
  }
  __syncthreads();
  const int N = 269;
  for (int i = t; i < 2 * N; i += 256) {
    int a = i / N, k = i - a * N;
    float re = 0.f, im = 0.f;
    for (int j = 0; j < 15; ++j) {
      float ang = -(PI2 * (float)(j * k)) / (float)N;
      float s, c; __sincosf(ang, &s, &c);
      re += p[a][j] * c; im += p[a][j] * s;
    }
    pw[a][k] = re * re + im * im;
  }
  __syncthreads();
  for (int i = t; i < 2 * N; i += 256) {
    int a = i / N, m = i - a * N;
    float s = 0.f;
    for (int k = 0; k < N; ++k) s += pw[a][k] * __cosf((PI2 * (float)(k * m)) / (float)N);
    zz[a][m] = s / (float)N;
  }
  __syncthreads();
  if (t < 2) {
    float m = -1e30f;
    for (int k = 0; k < N; ++k) m = fmaxf(m, zz[t][k]);
    mx[t] = m;
  }
  __syncthreads();
  for (int i = t; i < 2 * HP; i += 256) {
    int a = i / HP, m = i - a * HP;
    float zv = zz[a][(m == N) ? 0 : m];
    float v = 1.f - zv / mx[a];
    (a == 0 ? ah : aw)[m] = v;
  }
}

// ---------------------------------------------------------------------------
// OTF of 15x15 blur kernel (psf2otf, roll -7): direct tap evaluation
// ---------------------------------------------------------------------------
__global__ __launch_bounds__(256) void k_hf(const float* __restrict__ bk,
                                            float2* __restrict__ Hf) {
  int idx = blockIdx.x * 256 + threadIdx.x;
  if (idx >= PLANE) return;
  int u = idx / HP, v = idx - (idx / HP) * HP;
  float re = 0.f, im = 0.f;
  for (int i = 0; i < 15; ++i)
    for (int j = 0; j < 15; ++j) {
      float ang = -(PI2 / (float)HP) * (float)(u * (i - 7) + v * (j - 7));
      float s, c; __sincosf(ang, &s, &c);
      float w = bk[i * 15 + j];
      re += w * c; im += w * s;
    }
  Hf[idx] = make_float2(re, im);
}

// ---------------------------------------------------------------------------
// S = sum_f |Gf|^2 from 24 normalized 5x5 Wiener filters (roll -2)
// ---------------------------------------------------------------------------
__global__ __launch_bounds__(256) void k_S(const float* __restrict__ wwn,
                                           float* __restrict__ S) {
  int idx = blockIdx.x * 256 + threadIdx.x;
  if (idx >= PLANE) return;
  int u = idx / HP, v = idx - (idx / HP) * HP;
  float acc = 0.f;
  for (int f = 0; f < 24; ++f) {
    float re = 0.f, im = 0.f;
    for (int t = 0; t < 25; ++t) {
      int i = t / 5, j = t - (t / 5) * 5;
      float ang = -(PI2 / (float)HP) * (float)(u * (i - 2) + v * (j - 2));
      float s, c; __sincosf(ang, &s, &c);
      float w = wwn[f * 25 + t];
      re += w * c; im += w * s;
    }
    acc += re * re + im * im;
  }
  S[idx] = acc;
}

// ---------------------------------------------------------------------------
// Weight normalization: out = (w - mean)/||w - mean|| * scale  per channel
// ---------------------------------------------------------------------------
__global__ __launch_bounds__(256) void k_wnorm(const float* __restrict__ w,
                                               const float* __restrict__ scale,
                                               float* __restrict__ out, int fan) {
  __shared__ float s1s[256], s2s[256];
  int ch = blockIdx.x;
  const float* wc = w + (size_t)ch * fan;
  float s1 = 0.f, s2 = 0.f;
  for (int i = threadIdx.x; i < fan; i += 256) { float v = wc[i]; s1 += v; s2 += v * v; }
  s1s[threadIdx.x] = s1; s2s[threadIdx.x] = s2;
  __syncthreads();
  for (int o = 128; o > 0; o >>= 1) {
    if (threadIdx.x < o) { s1s[threadIdx.x] += s1s[threadIdx.x + o];
                           s2s[threadIdx.x] += s2s[threadIdx.x + o]; }
    __syncthreads();
  }
  float mean = s1s[0] / (float)fan;
  float nsq  = s2s[0] - s1s[0] * s1s[0] / (float)fan;
  float inv  = scale[ch] * rsqrtf(fmaxf(nsq, 1e-30f));
  for (int i = threadIdx.x; i < fan; i += 256)
    out[(size_t)ch * fan + i] = (wc[i] - mean) * inv;
}

// ---------------------------------------------------------------------------
// Naive DFT passes (row = along W, col = along H), complex, P planes
// ---------------------------------------------------------------------------
__global__ __launch_bounds__(256) void k_r2c(const float* __restrict__ in,
                                             float2* __restrict__ out, int n) {
  int i = blockIdx.x * 256 + threadIdx.x;
  if (i < n) out[i] = make_float2(in[i], 0.f);
}

__global__ __launch_bounds__(256) void k_dft_row(const float2* __restrict__ in,
                                                 float2* __restrict__ out,
                                                 int planes, float sign) {
  int idx = blockIdx.x * 256 + threadIdx.x;
  if (idx >= planes * PLANE) return;
  int p = idx / PLANE, r = idx - p * PLANE;
  int h = r / HP, v = r - h * HP;
  const float2* row = in + (size_t)p * PLANE + (size_t)h * HP;
  float step = sign * (PI2 * (float)v / (float)HP);
  float re = 0.f, im = 0.f;
  for (int w = 0; w < HP; ++w) {
    float s, c; __sincosf(step * (float)w, &s, &c);
    float2 x = row[w];
    re += x.x * c - x.y * s;
    im += x.x * s + x.y * c;
  }
  out[idx] = make_float2(re, im);
}

__global__ __launch_bounds__(256) void k_dft_col(const float2* __restrict__ in,
                                                 float2* __restrict__ out,
                                                 int planes, float sign, float scale) {
  int idx = blockIdx.x * 256 + threadIdx.x;
  if (idx >= planes * PLANE) return;
  int p = idx / PLANE, r = idx - p * PLANE;
  int u = r / HP, w = r - u * HP;
  const float2* col = in + (size_t)p * PLANE + w;
  float step = sign * (PI2 * (float)u / (float)HP);
  float re = 0.f, im = 0.f;
  for (int h = 0; h < HP; ++h) {
    float s, c; __sincosf(step * (float)h, &s, &c);
    float2 x = col[(size_t)h * HP];
    re += x.x * c - x.y * s;
    im += x.x * s + x.y * c;
  }
  out[idx] = make_float2(re * scale, im * scale);
}

__global__ __launch_bounds__(256) void k_mulHf(const float2* __restrict__ in,
                                               const float2* __restrict__ Hf,
                                               float2* __restrict__ out, int planes) {
  int idx = blockIdx.x * 256 + threadIdx.x;
  if (idx >= planes * PLANE) return;
  out[idx] = cmulc(in[idx], Hf[idx % PLANE]);
}

__global__ __launch_bounds__(256) void k_etcomb(float* __restrict__ xp,
                                                const float2* __restrict__ blur,
                                                const float* __restrict__ ah,
                                                const float* __restrict__ aw) {
  int idx = blockIdx.x * 256 + threadIdx.x;
  if (idx >= 2 * PLANE) return;
  int r = idx % PLANE;
  int h = r / HP, w = r - h * HP;
  float al = ah[h] * aw[w];
  xp[idx] = al * xp[idx] + (1.f - al) * blur[idx].x;
}

// T[d] * Y[b] -> 8 planes  (T = conj(Hf)/(|Hf|^2 + a_d S))
__global__ __launch_bounds__(256) void k_applyT(const float2* __restrict__ Y,
                                                const float2* __restrict__ Hf,
                                                const float* __restrict__ S,
                                                const float* __restrict__ alpha,
                                                float2* __restrict__ out) {
  int idx = blockIdx.x * 256 + threadIdx.x;
  if (idx >= 4 * PLANE) return;
  int d = idx / PLANE, r = idx - d * PLANE;
  float2 H = Hf[r];
  float a = __expf(alpha[d]);
  float den = H.x * H.x + H.y * H.y + a * S[r];
  float inv = 1.f / den;
  float2 T = make_float2(H.x * inv, -H.y * inv);
  for (int b = 0; b < 2; ++b) {
    float2 y = Y[(size_t)b * PLANE + r];
    out[((size_t)(b * 4 + d)) * PLANE + r] = cmulc(T, y);
  }
}

__global__ __launch_bounds__(256) void k_takereal(const float2* __restrict__ in,
                                                  float* __restrict__ out, int n) {
  int i = blockIdx.x * 256 + threadIdx.x;
  if (i < n) out[i] = in[i].x;
}

// amp-mean per d, then eps[8]
__global__ __launch_bounds__(256) void k_ampeps(const float2* __restrict__ Hf,
                                                const float* __restrict__ S,
                                                const float* __restrict__ alpha,
                                                const float* __restrict__ stdn,
                                                const float* __restrict__ aproj,
                                                float* __restrict__ eps) {
  __shared__ float red[256];
  __shared__ float amp[4];
  int t = threadIdx.x;
  for (int d = 0; d < 4; ++d) {
    float a = __expf(alpha[d]);
    float s = 0.f;
    for (int i = t; i < PLANE; i += 256) {
      float2 H = Hf[i];
      float m2 = H.x * H.x + H.y * H.y;
      float den = m2 + a * S[i];
      s += m2 / (den * den);
    }
    red[t] = s; __syncthreads();
    for (int o = 128; o > 0; o >>= 1) {
      if (t < o) red[t] += red[t + o];
      __syncthreads();
    }
    if (t == 0) amp[d] = red[0] / (float)PLANE;
    __syncthreads();
  }
  if (t < 8) {
    int b = t / 4, d = t - (t / 4) * 4;
    float cst = sqrtf(stdn[b] * stdn[b] * amp[d]);
    eps[t] = __expf(aproj[0]) * cst * sqrtf((float)(PLANE - 1));
  }
}

// ---------------------------------------------------------------------------
// Front conv 5x5 1->32 (symmetric pad 2) + bias
// ---------------------------------------------------------------------------
__global__ __launch_bounds__(256) void k_conv5(const float* __restrict__ z,
                                               const float* __restrict__ cw,
                                               const float* __restrict__ bias,
                                               float* __restrict__ out) {
  __shared__ float wsh[800];
  __shared__ float bsh[32];
  for (int i = threadIdx.x; i < 800; i += 256) wsh[i] = cw[i];
  if (threadIdx.x < 32) bsh[threadIdx.x] = bias[threadIdx.x];
  __syncthreads();
  int idx = blockIdx.x * 256 + threadIdx.x;
  if (idx >= 8 * PLANE) return;
  int img = idx / PLANE, r = idx - img * PLANE;
  int h = r / HP, w = r - h * HP;
  float patch[25];
  for (int u = 0; u < 5; ++u)
    for (int v = 0; v < 5; ++v) {
      int rr = reflectI(h + u - 2, HP), cc = reflectI(w + v - 2, HP);
      patch[u * 5 + v] = z[(size_t)img * PLANE + rr * HP + cc];
    }
  for (int c = 0; c < 32; ++c) {
    float acc = bsh[c];
#pragma unroll
    for (int t = 0; t < 25; ++t) acc += patch[t] * wsh[c * 25 + t];
    out[((size_t)img * 32 + c) * PLANE + r] = acc;
  }
}

// ---------------------------------------------------------------------------
// WMMA implicit-GEMM 3x3 conv (symmetric pad 1), fused PReLU on input, bias,
// optional residual add.  M = COUT, K = CIN*9, N = 16-pixel row tiles.
// LDS fragments pre-packed per lane, contiguous k = 16*(lane/16) + element for A and B.
// ---------------------------------------------------------------------------
template <int CIN, int COUT, int NT>
__global__ __launch_bounds__(256) void k_conv3x3_wmma(
    const float* __restrict__ in, const float* __restrict__ w,
    const float* __restrict__ bias, const float* __restrict__ slope,
    const float* __restrict__ resid, float* __restrict__ out) {
  constexpr int KT  = (CIN * 9) / 32;   // total k-tiles of 32
  constexpr int KTS = 9;                // k-tiles per LDS stage
  constexpr int KST = KT / KTS;         // stages (1 or 2)
  constexpr int MT  = COUT / 16;
  constexpr int NPX = NT * 16;
  static_assert(MT * NT == 8, "one wave per C tile");
  static_assert(KT % KTS == 0, "stage split");

  __shared__ uint4 aF[MT * KTS * 32 * 2];
  __shared__ uint4 bF[NT * KTS * 32 * 2];

  const int img = blockIdx.z;
  const int h   = blockIdx.y;
  const int c0  = blockIdx.x * NPX;
  const int tid = threadIdx.x;

  const float* inp = in + (size_t)img * CIN * PLANE;
  __builtin_prefetch(inp + (size_t)h * HP, 0, 1);

  const int wv   = tid >> 5;
  const int lane = tid & 31;
  const int mt   = wv / NT;
  const int nt   = wv - mt * NT;

  v8f acc = {};

  for (int st = 0; st < KST; ++st) {
    if (st) __syncthreads();
    // --- stage A fragments (weights -> f16, A k = 16*(lane/16) + elem) ---
    for (int s = tid; s < MT * KTS * 32 * 2; s += 256) {
      int q = s & 1;
      int ln = (s >> 1) & 31;
      int f = s >> 6;
      int kt = f % KTS, mtf = f / KTS;
      int m  = mtf * 16 + (ln & 15);
      int hi = ln >> 4;
      uint4 val;
      u32 pk[4];
#pragma unroll
      for (int j = 0; j < 4; ++j) {
        int v  = q * 4 + j;
        int kl = (hi << 4) + (v << 1);
        int k  = (st * KTS + kt) * 32 + kl;
        float w0 = w[(size_t)m * (CIN * 9) + k];
        float w1 = w[(size_t)m * (CIN * 9) + k + 1];
        pk[j] = packh2(w0, w1);
      }
      val.x = pk[0]; val.y = pk[1]; val.z = pk[2]; val.w = pk[3];
      aF[s] = val;
    }
    // --- stage B fragments (PReLU'd im2col -> f16, B k = 16*(lane/16) + elem) ---
    for (int s = tid; s < NT * KTS * 32 * 2; s += 256) {
      int q = s & 1;
      int ln = (s >> 1) & 31;
      int f = s >> 6;
      int kt = f % KTS, ntf = f / KTS;
      int col = c0 + ntf * 16 + (ln & 15);
      int hi = ln >> 4;
      uint4 val;
      u32 pk[4];
#pragma unroll
      for (int j = 0; j < 4; ++j) {
        int v  = q * 4 + j;
        int kl = (hi << 4) + (v << 1);
        int k  = (st * KTS + kt) * 32 + kl;
        float x0 = 0.f, x1 = 0.f;
        if (col < HP) {
          int cin = k / 9, tap = k - cin * 9;
          int rr = reflectI(h + tap / 3 - 1, HP);
          int cc = reflectI(col + (tap - (tap / 3) * 3) - 1, HP);
          float xv = inp[(size_t)cin * PLANE + rr * HP + cc];
          x0 = (xv >= 0.f) ? xv : slope[cin] * xv;
          int k1 = k + 1;
          cin = k1 / 9; tap = k1 - cin * 9;
          rr = reflectI(h + tap / 3 - 1, HP);
          cc = reflectI(col + (tap - (tap / 3) * 3) - 1, HP);
          xv = inp[(size_t)cin * PLANE + rr * HP + cc];
          x1 = (xv >= 0.f) ? xv : slope[cin] * xv;
        }
        pk[j] = packh2(x0, x1);
      }
      val.x = pk[0]; val.y = pk[1]; val.z = pk[2]; val.w = pk[3];
      bF[s] = val;
    }
    __syncthreads();
    // --- WMMA over this stage's k-tiles ---
#pragma unroll
    for (int ktl = 0; ktl < KTS; ++ktl) {
      const uint4* ap = &aF[((mt * KTS + ktl) * 32 + lane) * 2];
      const uint4* bp = &bF[((nt * KTS + ktl) * 32 + lane) * 2];
      v16h av, bv;
      ((uint4*)&av)[0] = ap[0]; ((uint4*)&av)[1] = ap[1];
      ((uint4*)&bv)[0] = bp[0]; ((uint4*)&bv)[1] = bp[1];
      acc = __builtin_amdgcn_wmma_f32_16x16x32_f16(
          false, av, false, bv, (short)0, acc, false, false);
    }
  }

  // --- store C (ISA C layout: VGPR r -> M = r + 8*(lane>=16), N = lane&15) ---
  int col = c0 + nt * 16 + (lane & 15);
  if (col < HP) {
    int mbase = mt * 16 + ((lane >> 4) << 3);
#pragma unroll
    for (int r = 0; r < 8; ++r) {
      int m = mbase + r;
      float val = acc[r] + bias[m];
      size_t oi = ((size_t)img * COUT + m) * PLANE + (size_t)h * HP + col;
      if (resid) val += resid[oi];
      out[oi] = val;
    }
  }
}

// ---------------------------------------------------------------------------
// Transposed conv 5x5 32->1 (zero pad, crop 2) + bias_t
// ---------------------------------------------------------------------------
__global__ __launch_bounds__(256) void k_convT5(const float* __restrict__ feat,
                                                const float* __restrict__ cw,
                                                const float* __restrict__ bias_t,
                                                float* __restrict__ out) {
  __shared__ float wsh[800];
  for (int i = threadIdx.x; i < 800; i += 256) wsh[i] = cw[i];
  __syncthreads();
  int idx = blockIdx.x * 256 + threadIdx.x;
  if (idx >= 8 * PLANE) return;
  int img = idx / PLANE, r = idx - img * PLANE;
  int h = r / HP, w = r - h * HP;
  float acc = bias_t[0];
  for (int c = 0; c < 32; ++c) {
    const float* fp = feat + ((size_t)img * 32 + c) * PLANE;
    for (int u = 0; u < 5; ++u) {
      int rr = h + u - 2;
      if (rr < 0 || rr >= HP) continue;
      for (int v = 0; v < 5; ++v) {
        int cc = w + v - 2;
        if (cc < 0 || cc >= HP) continue;
        acc += fp[(size_t)rr * HP + cc] * wsh[c * 25 + (4 - u) * 5 + (4 - v)];
      }
    }
  }
  out[idx] = acc;
}

// per-plane L2 norm of convT output
__global__ __launch_bounds__(256) void k_norm(const float* __restrict__ oT,
                                              float* __restrict__ nrm) {
  __shared__ float red[256];
  int n = blockIdx.x;
  float s = 0.f;
  for (int i = threadIdx.x; i < PLANE; i += 256) {
    float v = oT[(size_t)n * PLANE + i];
    s += v * v;
  }
  red[threadIdx.x] = s; __syncthreads();
  for (int o = 128; o > 0; o >>= 1) {
    if (threadIdx.x < o) red[threadIdx.x] += red[threadIdx.x + o];
    __syncthreads();
  }
  if (threadIdx.x == 0) nrm[n] = sqrtf(red[0]);
}

// projection + clip + crop + mix over D
__global__ __launch_bounds__(256) void k_final(const float* __restrict__ z,
                                               const float* __restrict__ oT,
                                               const float* __restrict__ nrm,
                                               const float* __restrict__ eps,
                                               const float* __restrict__ mix,
                                               float* __restrict__ out) {
  int idx = blockIdx.x * 256 + threadIdx.x;
  if (idx >= 2 * 65536) return;
  int b = idx / 65536, r = idx - b * 65536;
  int oh = r / 256, ow = r - oh * 256;
  float acc = 0.f;
  for (int d = 0; d < 4; ++d) {
    int n = b * 4 + d;
    float nr = nrm[n], ep = eps[n];
    float s = (nr > ep) ? (ep / fmaxf(nr, 1e-12f)) : 1.f;
    size_t pi = (size_t)n * PLANE + (size_t)(oh + 7) * HP + (ow + 7);
    float v = z[pi] - oT[pi] * s;
    v = fminf(fmaxf(v, 0.f), 255.f);
    acc += mix[d] * v;
  }
  out[idx] = acc;
}

// ---------------------------------------------------------------------------
// Orchestration
// ---------------------------------------------------------------------------
static inline int LB(long n) { return (int)((n + 255) / 256); }

extern "C" void kernel_launch(void* const* d_in, const int* in_sizes, int n_in,
                              void* d_out, int out_size, void* d_ws, size_t ws_size,
                              hipStream_t stream) {
  (void)in_sizes; (void)n_in; (void)out_size; (void)ws_size;
  const float* x        = (const float*)d_in[0];
  const float* bk       = (const float*)d_in[1];
  const float* stdn     = (const float*)d_in[2];
  const float* wiener_w = (const float*)d_in[3];
  const float* wiener_s = (const float*)d_in[4];
  const float* alpha    = (const float*)d_in[5];
  const float* conv_w   = (const float*)d_in[6];
  const float* scale_f  = (const float*)d_in[7];
  const float* bias_f   = (const float*)d_in[8];
  const float* bias_t   = (const float*)d_in[9];
  const float* prelu1   = (const float*)d_in[10];
  const float* w1       = (const float*)d_in[11];
  const float* scale1   = (const float*)d_in[12];
  const float* b1       = (const float*)d_in[13];
  const float* prelu2   = (const float*)d_in[14];
  const float* w2       = (const float*)d_in[15];
  const float* scale2   = (const float*)d_in[16];
  const float* b2       = (const float*)d_in[17];
  const float* aproj    = (const float*)d_in[18];
  const float* mixw     = (const float*)d_in[19];
  float* out = (float*)d_out;

  size_t off = 0;
  auto alloc = [&](size_t nf) -> char* {
    char* p = (char*)d_ws + off;
    off += ((nf * sizeof(float) + 255) / 256) * 256;
    return p;
  };
  float*  xp  = (float*) alloc((size_t)2 * PLANE);
  float2* C1  = (float2*)alloc((size_t)8 * PLANE * 2);
  float2* C2  = (float2*)alloc((size_t)8 * PLANE * 2);
  float2* Hf  = (float2*)alloc((size_t)PLANE * 2);
  float*  Sb  = (float*) alloc((size_t)PLANE);
  float*  ah  = (float*) alloc(HP);
  float*  aw  = (float*) alloc(HP);
  float*  wwn = (float*) alloc(24 * 25);
  float*  cwn = (float*) alloc(32 * 25);
  float*  w1n = (float*) alloc((size_t)5 * 64 * 32 * 9);
  float*  w2n = (float*) alloc((size_t)5 * 32 * 64 * 9);
  float*  z   = (float*) alloc((size_t)8 * PLANE);
  float*  A32 = (float*) alloc((size_t)8 * 32 * PLANE);
  float*  B32 = (float*) alloc((size_t)8 * 32 * PLANE);
  float*  T64 = (float*) alloc((size_t)8 * 64 * PLANE);
  float*  oT  = (float*) alloc((size_t)8 * PLANE);
  float*  eps = (float*) alloc(8);
  float*  nrm = (float*) alloc(8);

  // --- setup: pad, edgetaper profiles, OTFs, weight norms ---
  k_sympad<<<LB(2 * PLANE), 256, 0, stream>>>(x, xp);
  k_et1d<<<1, 256, 0, stream>>>(bk, ah, aw);
  k_hf<<<LB(PLANE), 256, 0, stream>>>(bk, Hf);
  k_wnorm<<<24, 256, 0, stream>>>(wiener_w, wiener_s, wwn, 25);
  k_wnorm<<<32, 256, 0, stream>>>(conv_w, scale_f, cwn, 25);
  for (int i = 0; i < 5; ++i) {
    k_wnorm<<<64, 256, 0, stream>>>(w1 + (size_t)i * 64 * 288, scale1 + i * 64,
                                    w1n + (size_t)i * 64 * 288, 288);
    k_wnorm<<<32, 256, 0, stream>>>(w2 + (size_t)i * 32 * 576, scale2 + i * 32,
                                    w2n + (size_t)i * 32 * 576, 576);
  }
  k_S<<<LB(PLANE), 256, 0, stream>>>(wwn, Sb);

  // --- edgetaper: xp = a*xp + (1-a)*ifft2(fft2(xp)*Hf).real ---
  k_r2c<<<LB(2 * PLANE), 256, 0, stream>>>(xp, C1, 2 * PLANE);
  k_dft_row<<<LB(2 * PLANE), 256, 0, stream>>>(C1, C2, 2, -1.f);
  k_dft_col<<<LB(2 * PLANE), 256, 0, stream>>>(C2, C1, 2, -1.f, 1.f);
  k_mulHf<<<LB(2 * PLANE), 256, 0, stream>>>(C1, Hf, C2, 2);
  k_dft_row<<<LB(2 * PLANE), 256, 0, stream>>>(C2, C1, 2, 1.f);
  k_dft_col<<<LB(2 * PLANE), 256, 0, stream>>>(C1, C2, 2, 1.f, 1.f / (float)PLANE);
  k_etcomb<<<LB(2 * PLANE), 256, 0, stream>>>(xp, C2, ah, aw);

  // --- Wiener filtering -> z (8 planes) ---
  k_r2c<<<LB(2 * PLANE), 256, 0, stream>>>(xp, C1, 2 * PLANE);
  k_dft_row<<<LB(2 * PLANE), 256, 0, stream>>>(C1, C2, 2, -1.f);
  k_dft_col<<<LB(2 * PLANE), 256, 0, stream>>>(C2, C1, 2, -1.f, 1.f);   // C1 = Y
  k_applyT<<<LB(4 * PLANE), 256, 0, stream>>>(C1, Hf, Sb, alpha, C2);   // 8 planes
  k_dft_row<<<LB(8 * PLANE), 256, 0, stream>>>(C2, C1, 8, 1.f);
  k_dft_col<<<LB(8 * PLANE), 256, 0, stream>>>(C1, C2, 8, 1.f, 1.f / (float)PLANE);
  k_takereal<<<LB(8 * PLANE), 256, 0, stream>>>(C2, z, 8 * PLANE);
  k_ampeps<<<1, 256, 0, stream>>>(Hf, Sb, alpha, stdn, aproj, eps);

  // --- CNN ---
  k_conv5<<<LB(8 * PLANE), 256, 0, stream>>>(z, cwn, bias_f, A32);
  float* cur = A32;
  float* nxt = B32;
  for (int i = 0; i < 5; ++i) {
    // 32 -> 64 : MT=4, NT=2, 32 px/block
    k_conv3x3_wmma<32, 64, 2><<<dim3((HP + 31) / 32, HP, 8), 256, 0, stream>>>(
        cur, w1n + (size_t)i * 64 * 288, b1 + i * 64, prelu1 + i * 32, nullptr, T64);
    // 64 -> 32 : MT=2, NT=4, 64 px/block, residual add on block 0
    k_conv3x3_wmma<64, 32, 4><<<dim3((HP + 63) / 64, HP, 8), 256, 0, stream>>>(
        T64, w2n + (size_t)i * 32 * 576, b2 + i * 32, prelu2 + i * 64,
        (i == 0) ? cur : nullptr, nxt);
    float* t = cur; cur = nxt; nxt = t;
  }

  // --- transposed conv, norm projection, clip/crop/mix ---
  k_convT5<<<LB(8 * PLANE), 256, 0, stream>>>(cur, cwn, bias_t, oT);
  k_norm<<<8, 256, 0, stream>>>(oT, nrm);
  k_final<<<LB(2 * 65536), 256, 0, stream>>>(z, oT, nrm, eps, mixw, out);
}
